// MultiHeadAttention_54820962566281
// MI455X (gfx1250) — matrix operand order repaired and checked by reading, not compiled, not run on hardware
//
#include <hip/hip_runtime.h>

// ---------------------------------------------------------------------------
// MultiHeadAttention forward for MI455X (gfx1250, wave32, WMMA bf16)
// B=2, T=2048, D=1024, H=16, Hd=64.  Outputs: out[B,T,D], k[B,H,T,Hd], v[B,H,T,Hd]
// ---------------------------------------------------------------------------

typedef __attribute__((ext_vector_type(16))) __bf16      bf16x16;
typedef __attribute__((ext_vector_type(8)))  float       f32x8;
typedef __attribute__((ext_vector_type(4)))  unsigned int u32x4;

union Frag16 { bf16x16 bf; u32x4 q[2]; };

__device__ __forceinline__ unsigned short f2bf(float f) {
  unsigned int u = __builtin_bit_cast(unsigned int, f);
  u += 0x7FFFu + ((u >> 16) & 1u);          // round-to-nearest-even
  return (unsigned short)(u >> 16);
}

// Load a 16-element bf16 fragment from two 16-byte chunks.
__device__ __forceinline__ bf16x16 load_frag2(const unsigned short* p0,
                                              const unsigned short* p1) {
  Frag16 f;
  f.q[0] = *(const u32x4*)p0;
  f.q[1] = *(const u32x4*)p1;
  return f.bf;
}

__device__ __forceinline__ f32x8 wmma_bf16(bf16x16 a, bf16x16 b, f32x8 c) {
  return __builtin_amdgcn_wmma_f32_16x16x32_bf16(
      /*neg_a=*/false, a, /*neg_b=*/false, b,
      /*c_mod=*/(short)0, c, /*reuse_a=*/false, /*reuse_b=*/false);
}

// ---------------------------------------------------------------------------
// f32 -> bf16 cast
// ---------------------------------------------------------------------------
__global__ void cast_kernel(const float* __restrict__ src,
                            unsigned short* __restrict__ dst, int n) {
  for (int i = blockIdx.x * blockDim.x + threadIdx.x; i < n;
       i += gridDim.x * blockDim.x)
    dst[i] = f2bf(src[i]);
}

// ---------------------------------------------------------------------------
// Projection GEMM:  Y[m][n] = sum_k X[m][k] * W[n][k] + bias[n]
// M=4096, N=1024, K=1024. Each wave computes a 32x64 tile (2 A x 4 B frags).
// mode 0: Q  -> obf  [B,H,T,Hd]
// mode 1: K  -> obf  [B,H,T,Hd]  + of32 [B,H,T,Hd]
// mode 2: V  -> obf  [B,H,Hd,T]  + of32 [B,H,T,Hd]
// mode 3: O  ->                    of32 [B*T, D]
// ---------------------------------------------------------------------------
__global__ __launch_bounds__(256) void proj_gemm(
    const unsigned short* __restrict__ X,
    const unsigned short* __restrict__ W,
    const float* __restrict__ bias,
    unsigned short* __restrict__ obf,
    float* __restrict__ of32,
    int mode) {
  const int lane = threadIdx.x & 31;
  const int wid  = threadIdx.x >> 5;
  const int wg   = blockIdx.x * 8 + wid;     // 2048 wave tiles
  const int nt   = wg & 15;                  // 16 N tiles of 64
  const int mt   = wg >> 4;                  // 128 M tiles of 32
  const int m0   = mt * 32, n0 = nt * 64;
  const int half = lane >> 4, l15 = lane & 15;

  f32x8 c[2][4];
#pragma unroll
  for (int a = 0; a < 2; ++a)
#pragma unroll
    for (int j = 0; j < 4; ++j)
#pragma unroll
      for (int e = 0; e < 8; ++e) c[a][j][e] = 0.0f;

  for (int k0 = 0; k0 < 1024; k0 += 32) {
    const unsigned short* ap0 = X + (size_t)(m0 + l15) * 1024 + k0 + half * 8;
    const unsigned short* ap1 = X + (size_t)(m0 + 16 + l15) * 1024 + k0 + half * 8;
    bf16x16 a0 = load_frag2(ap0, ap0 + 16);
    bf16x16 a1 = load_frag2(ap1, ap1 + 16);
#pragma unroll
    for (int j = 0; j < 4; ++j) {
      const unsigned short* bp =
          W + (size_t)(n0 + j * 16 + l15) * 1024 + k0 + half * 8;
      bf16x16 b = load_frag2(bp, bp + 16);
      c[0][j] = wmma_bf16(a0, b, c[0][j]);
      c[1][j] = wmma_bf16(a1, b, c[1][j]);
    }
  }

#pragma unroll
  for (int a = 0; a < 2; ++a) {
#pragma unroll
    for (int j = 0; j < 4; ++j) {
      const int col = n0 + j * 16 + l15;
      const float bv = bias[col];
      const int h = col >> 6, d = col & 63;
#pragma unroll
      for (int e = 0; e < 8; ++e) {
        const int row = m0 + a * 16 + half * 8 + e;   // global token index
        const int bi = row >> 11, t = row & 2047;
        const float val = c[a][j][e] + bv;
        if (mode == 0) {
          obf[((size_t)(bi * 16 + h) * 2048 + t) * 64 + d] = f2bf(val);
        } else if (mode == 1) {
          const size_t idx = ((size_t)(bi * 16 + h) * 2048 + t) * 64 + d;
          obf[idx]  = f2bf(val);
          of32[idx] = val;
        } else if (mode == 2) {
          obf[((size_t)(bi * 16 + h) * 64 + d) * 2048 + t] = f2bf(val);
          of32[((size_t)(bi * 16 + h) * 2048 + t) * 64 + d] = val;
        } else {
          of32[(size_t)row * 1024 + col] = val;
        }
      }
    }
  }
}

// ---------------------------------------------------------------------------
// Flash attention, causal. One wave per 16-row Q tile. K-blocks of 32.
// Q,K: [B,H,T,Hd] bf16.  Vt: [B,H,Hd,T] bf16.  ctx out: [B*T, H*Hd] bf16.
// ---------------------------------------------------------------------------
__global__ __launch_bounds__(256) void attn_kernel(
    const unsigned short* __restrict__ Q,
    const unsigned short* __restrict__ K,
    const unsigned short* __restrict__ Vt,
    unsigned short* __restrict__ ctx) {
  __shared__ unsigned short pbuf[8][16 * 32];   // per-wave P tile (16x32 bf16)
  const int lane = threadIdx.x & 31;
  const int wid  = threadIdx.x >> 5;
  const int wg   = blockIdx.x * 8 + wid;        // 4096 wave tiles
  const int bh   = wg >> 7;                     // (b*16+h)
  const int q0   = (wg & 127) * 16;
  const int half = lane >> 4, l15 = lane & 15;

  const unsigned short* Qb = Q  + (size_t)bh * 2048 * 64;
  const unsigned short* Kb = K  + (size_t)bh * 2048 * 64;
  const unsigned short* Vb = Vt + (size_t)bh * 64 * 2048;

  bf16x16 qf[2];
#pragma unroll
  for (int dd = 0; dd < 2; ++dd) {
    const unsigned short* p = Qb + (size_t)(q0 + l15) * 64 + dd * 32 + half * 8;
    qf[dd] = load_frag2(p, p + 16);
  }

  f32x8 o[4];
  float mi[8], li[8];
#pragma unroll
  for (int j = 0; j < 4; ++j)
#pragma unroll
    for (int e = 0; e < 8; ++e) o[j][e] = 0.0f;
#pragma unroll
  for (int e = 0; e < 8; ++e) { mi[e] = -1e30f; li[e] = 0.0f; }

  unsigned short* pb = pbuf[wid];

  for (int kb = 0; kb < q0 + 16; kb += 32) {
    // S = Q @ K^T for 32 key columns (two 16-wide tiles), k-dim = 64
    f32x8 s0, s1;
#pragma unroll
    for (int e = 0; e < 8; ++e) { s0[e] = 0.0f; s1[e] = 0.0f; }
#pragma unroll
    for (int dd = 0; dd < 2; ++dd) {
      const unsigned short* p0 =
          Kb + (size_t)(kb + l15) * 64 + dd * 32 + half * 8;
      const unsigned short* p1 =
          Kb + (size_t)(kb + 16 + l15) * 64 + dd * 32 + half * 8;
      bf16x16 kf0 = load_frag2(p0, p0 + 16);
      bf16x16 kf1 = load_frag2(p1, p1 + 16);
      s0 = wmma_bf16(qf[dd], kf0, s0);
      s1 = wmma_bf16(qf[dd], kf1, s1);
    }

    // online softmax (row = q0 + half*8 + e; col = kb + {0,16} + l15)
    float alpha[8];
#pragma unroll
    for (int e = 0; e < 8; ++e) {
      const int row = q0 + half * 8 + e;
      float v0 = s0[e] * 0.125f;                 // 1/sqrt(64)
      float v1 = s1[e] * 0.125f;
      if (kb + l15 > row)      v0 = -1e30f;      // causal mask
      if (kb + 16 + l15 > row) v1 = -1e30f;
      float m = fmaxf(v0, v1);
      m = fmaxf(m, __shfl_xor(m, 1, 32));
      m = fmaxf(m, __shfl_xor(m, 2, 32));
      m = fmaxf(m, __shfl_xor(m, 4, 32));
      m = fmaxf(m, __shfl_xor(m, 8, 32));
      const float nm = fmaxf(mi[e], m);
      const float al = __expf(mi[e] - nm);
      const float p0 = __expf(v0 - nm);
      const float p1 = __expf(v1 - nm);
      s0[e] = p0; s1[e] = p1;
      float rs = p0 + p1;
      rs += __shfl_xor(rs, 1, 32);
      rs += __shfl_xor(rs, 2, 32);
      rs += __shfl_xor(rs, 4, 32);
      rs += __shfl_xor(rs, 8, 32);
      li[e] = li[e] * al + rs;
      mi[e] = nm;
      alpha[e] = al;
    }
#pragma unroll
    for (int j = 0; j < 4; ++j)
#pragma unroll
      for (int e = 0; e < 8; ++e) o[j][e] *= alpha[e];

    // P (C-layout) -> bf16 A-fragment layout via per-wave LDS slice
#pragma unroll
    for (int e = 0; e < 8; ++e) {
      const int r = half * 8 + e;
      pb[r * 32 + l15]      = f2bf(s0[e]);
      pb[r * 32 + 16 + l15] = f2bf(s1[e]);
    }
    asm volatile("s_wait_dscnt 0x0" ::: "memory");
    const unsigned short* ap = pb + l15 * 32 + half * 8;
    bf16x16 pf = load_frag2(ap, ap + 16);

    // O += P @ V  (V fragments contiguous thanks to Vt layout)
#pragma unroll
    for (int j = 0; j < 4; ++j) {
      const unsigned short* vp =
          Vb + (size_t)(j * 16 + l15) * 2048 + kb + half * 8;
      bf16x16 vf = load_frag2(vp, vp + 16);
      o[j] = wmma_bf16(pf, vf, o[j]);
    }
    asm volatile("" ::: "memory");   // keep next-iter LDS stores after loads
  }

  // epilogue: normalize and store ctx[b*2048+row][h*64+col] as bf16
  const int b = bh >> 4, h = bh & 15;
#pragma unroll
  for (int j = 0; j < 4; ++j) {
    const int col = h * 64 + j * 16 + l15;
#pragma unroll
    for (int e = 0; e < 8; ++e) {
      const int row = q0 + half * 8 + e;
      ctx[(size_t)(b * 2048 + row) * 1024 + col] = f2bf(o[j][e] / li[e]);
    }
  }
}

// ---------------------------------------------------------------------------
extern "C" void kernel_launch(void* const* d_in, const int* in_sizes, int n_in,
                              void* d_out, int out_size, void* d_ws,
                              size_t ws_size, hipStream_t stream) {
  (void)in_sizes; (void)n_in; (void)out_size; (void)ws_size;
  const float* x  = (const float*)d_in[0];
  const float* Wq = (const float*)d_in[1];
  const float* bq = (const float*)d_in[2];
  const float* Wk = (const float*)d_in[3];
  const float* bk = (const float*)d_in[4];
  const float* Wv = (const float*)d_in[5];
  const float* bv = (const float*)d_in[6];
  const float* Wo = (const float*)d_in[7];
  const float* bo = (const float*)d_in[8];

  float* out  = (float*)d_out;                      // [2,2048,1024]
  float* kout = out  + (size_t)2 * 2048 * 1024;     // [2,16,2048,64]
  float* vout = kout + (size_t)2 * 2048 * 1024;     // [2,16,2048,64]

  const size_t MN = (size_t)4096 * 1024;   // 4.19M elems
  const size_t WW = (size_t)1024 * 1024;
  unsigned short* XB  = (unsigned short*)d_ws;  // x bf16 (reused later as CTX)
  unsigned short* WQB = XB  + MN;
  unsigned short* WKB = WQB + WW;
  unsigned short* WVB = WKB + WW;
  unsigned short* WOB = WVB + WW;
  unsigned short* QB  = WOB + WW;               // [B,H,T,Hd] bf16
  unsigned short* KB  = QB  + MN;               // [B,H,T,Hd] bf16
  unsigned short* VTB = KB  + MN;               // [B,H,Hd,T] bf16
  unsigned short* CTX = XB;                     // reuse: x dead after V proj

  cast_kernel<<<1024, 256, 0, stream>>>(x,  XB,  (int)MN);
  cast_kernel<<<512,  256, 0, stream>>>(Wq, WQB, (int)WW);
  cast_kernel<<<512,  256, 0, stream>>>(Wk, WKB, (int)WW);
  cast_kernel<<<512,  256, 0, stream>>>(Wv, WVB, (int)WW);
  cast_kernel<<<512,  256, 0, stream>>>(Wo, WOB, (int)WW);

  proj_gemm<<<256, 256, 0, stream>>>(XB, WQB, bq, QB,  nullptr, 0);
  proj_gemm<<<256, 256, 0, stream>>>(XB, WKB, bk, KB,  kout,    1);
  proj_gemm<<<256, 256, 0, stream>>>(XB, WVB, bv, VTB, vout,    2);
  attn_kernel<<<512, 256, 0, stream>>>(QB, KB, VTB, CTX);
  proj_gemm<<<256, 256, 0, stream>>>(CTX, WOB, bo, nullptr, out, 3);
}
